// GCNeXt_19284403159426
// MI455X (gfx1250) — matrix operand order repaired and checked by reading, not compiled, not run on hardware
//
#include <hip/hip_runtime.h>
#include <math.h>

// Problem constants (from reference)
static constexpr int BB = 4;
static constexpr int CC = 256;
static constexpr int TT = 2048;
static constexpr int KK = 7;      // knn k
static constexpr int LL = 2;      // layers
static constexpr int GG = 32;     // groups
static constexpr int CG = CC / GG; // 8 channels per group
static constexpr int NEL = BB * CC * TT; // elements of one [B,C,T] tensor

typedef __bf16 bf16_t;
typedef bf16_t v16bf __attribute__((ext_vector_type(16)));
typedef float  v8f   __attribute__((ext_vector_type(8)));

union V16BF { v16bf v; uint4 q[2]; };

__device__ __forceinline__ unsigned short f2bf(float f) {
    union { float f; unsigned int u; } x; x.f = f;
    unsigned int u = x.u;
    unsigned int r = (u + 0x7FFFu + ((u >> 16) & 1u)) >> 16; // RNE
    return (unsigned short)r;
}

__device__ __forceinline__ float gelu_exact(float x) {
    return 0.5f * x * (1.0f + erff(x * 0.70710678118654752f));
}

// ---- WMMA fragment loaders ------------------------------------------------
// A (16x32 bf16): lane l holds row M = l&15.  Elements e=0..7 -> K = e + 8*(l>>4),
// e=8..15 -> K = 16 + (e-8) + 8*(l>>4).  rowbase = mat + M*stride + k0.
__device__ __forceinline__ v16bf load_a_frag(const unsigned short* rowbase, int l8) {
    V16BF r;
    r.q[0] = *(const uint4*)(rowbase + l8);
    r.q[1] = *(const uint4*)(rowbase + 16 + l8);
    return r.v;
}
// B (32x16 bf16): lane l holds col N = l&15.  Elements e=0..7 -> K = e + 8*(l>>4),
// e=8..15 -> K = 16 + (e-8) + 8*(l>>4).
// rowbase = mat + N*stride + k0 (B stored row-major by N with K contiguous).
__device__ __forceinline__ v16bf load_b_frag(const unsigned short* rowbase, int l16) {
    V16BF r;
    int l8 = l16 >> 1;
    r.q[0] = *(const uint4*)(rowbase + l8);
    r.q[1] = *(const uint4*)(rowbase + 16 + l8);
    return r.v;
}

// ---- Kernel 1: transpose + convert x [B,C,T] f32 -> xt [B,T,C] bf16 --------
__global__ void k_transpose_bf16(const float* __restrict__ x, unsigned short* __restrict__ xt) {
    int i = blockIdx.x * blockDim.x + threadIdx.x; // over B*T*C, c fastest
    int c = i % CC;
    int t = (i / CC) % TT;
    int b = i / (CC * TT);
    xt[i] = f2bf(x[((size_t)b * CC + c) * TT + t]);
}

// ---- Kernel 2: squared norms xx[b,t] --------------------------------------
__global__ void k_norms(const float* __restrict__ x, float* __restrict__ xx) {
    int i = blockIdx.x * blockDim.x + threadIdx.x; // B*T
    if (i >= BB * TT) return;
    int t = i % TT, b = i / TT;
    float s = 0.f;
    for (int c = 0; c < CC; ++c) {
        float v = x[((size_t)b * CC + c) * TT + t];
        s += v * v;
    }
    xx[i] = s;
}

// ---- Kernel 3: convert per-layer weights to bf16 --------------------------
__global__ void k_wcvt(const float* __restrict__ tw1i, const float* __restrict__ dw1i,
                       unsigned short* __restrict__ wt1, unsigned short* __restrict__ wnb,
                       unsigned short* __restrict__ wctr) {
    int i = blockIdx.x * blockDim.x + threadIdx.x; // C*C
    int c = i % CC, o = i / CC;
    wt1[i]  = f2bf(tw1i[i]);                             // tw1 [C,C]
    wnb[i]  = f2bf(dw1i[(size_t)o * 2 * CC + c]);        // dw1 [C, 0:C]   (neighbor)
    wctr[i] = f2bf(dw1i[(size_t)o * 2 * CC + CC + c]);   // dw1 [C, C:2C]  (center)
}

// ---- Kernel 4: GEMM  Y[b,o,t] = W[o,:] . xt[b,t,:] (+bias)(+gelu) ---------
// block = 256 (8 waves); wave w is N-stationary: holds the full 16t x 256c
// B-block in 8 fragments (loaded once) and sweeps 4 o-tiles over it.
__global__ void k_gemm(const unsigned short* __restrict__ W,
                       const unsigned short* __restrict__ xt,
                       const float* __restrict__ bias,
                       float* __restrict__ Y, int applyGelu) {
    int lane = threadIdx.x & 31;
    int wv   = threadIdx.x >> 5;
    int t0 = blockIdx.x * 128 + wv * 16;
    int o0 = blockIdx.y * 64;
    int b  = blockIdx.z;
    const unsigned short* xtb = xt + (size_t)b * TT * CC;
    int l8  = (lane >> 4) * 8;
    int l16 = (lane >> 4) * 16;
    const unsigned short* xrow = xtb + (size_t)(t0 + (lane & 15)) * CC;

    v16bf bfr[8];
#pragma unroll
    for (int c = 0; c < 8; ++c) bfr[c] = load_b_frag(xrow + 32 * c, l16);

    int n = t0 + (lane & 15);
#pragma unroll
    for (int j = 0; j < 4; ++j) {
        const unsigned short* wrow = W + (size_t)(o0 + 16 * j + (lane & 15)) * CC;
        v8f acc = {};
#pragma unroll
        for (int c = 0; c < 8; ++c) {
            v16bf af = load_a_frag(wrow + 32 * c, l8);
            acc = __builtin_amdgcn_wmma_f32_16x16x32_bf16(false, af, false, bfr[c],
                                                          (short)0, acc, false, false);
        }
#pragma unroll
        for (int r = 0; r < 8; ++r) {
            int m = o0 + 16 * j + r + l8;
            float v = acc[r];
            if (bias) v += bias[m];
            if (applyGelu) v = gelu_exact(v);
            Y[((size_t)b * CC + m) * TT + n] = v;
        }
    }
}

// ---- Kernel 5: fused KNN (WMMA Gram tiles + per-query top-7) --------------
// block = 32 (one wave) owns 16 query rows; the full 16x256 query A-block is
// hoisted into registers, then candidates stream through 64 at a time.
// score(m,n) = 2*<x_m,x_n> - ||x_n||^2   (the -||x_m||^2 term is rank-invariant)
__global__ void k_knn(const unsigned short* __restrict__ xt,
                      const float* __restrict__ xx, int* __restrict__ idxo) {
    __shared__ float s[16][64];
    int lane = threadIdx.x;
    int t0 = blockIdx.x * 16;
    int b  = blockIdx.y;
    const unsigned short* xtb = xt + (size_t)b * TT * CC;
    const float* xxb = xx + (size_t)b * TT;
    int l8  = (lane >> 4) * 8;
    int l16 = (lane >> 4) * 16;
    const unsigned short* qrow = xtb + (size_t)(t0 + (lane & 15)) * CC;

    v16bf qfr[8];
#pragma unroll
    for (int c = 0; c < 8; ++c) qfr[c] = load_a_frag(qrow + 32 * c, l8);

    float bs[KK]; int bi[KK];
#pragma unroll
    for (int j = 0; j < KK; ++j) { bs[j] = -3.0e38f; bi[j] = 0; }

    for (int n0 = 0; n0 < TT; n0 += 64) {
#pragma unroll
        for (int j = 0; j < 4; ++j) {
            const unsigned short* crow = xtb + (size_t)(n0 + 16 * j + (lane & 15)) * CC;
            v8f acc = {};
#pragma unroll
            for (int c = 0; c < 8; ++c) {
                v16bf bf = load_b_frag(crow + 32 * c, l16);
                acc = __builtin_amdgcn_wmma_f32_16x16x32_bf16(false, qfr[c], false, bf,
                                                              (short)0, acc, false, false);
            }
#pragma unroll
            for (int r = 0; r < 8; ++r) s[r + l8][16 * j + (lane & 15)] = acc[r];
        }
        __syncthreads();
        if (lane < 16) {
            for (int j = 0; j < 64; ++j) {
                float sc = 2.0f * s[lane][j] - xxb[n0 + j];
                if (sc > bs[KK - 1]) {          // strict > keeps earlier-index ties ahead
                    int p = KK - 1;
                    while (p > 0 && bs[p - 1] < sc) {
                        bs[p] = bs[p - 1]; bi[p] = bi[p - 1]; --p;
                    }
                    bs[p] = sc; bi[p] = n0 + j;
                }
            }
        }
        __syncthreads();
    }
    if (lane < 16) {
        int t = t0 + lane;
#pragma unroll
        for (int j = 0; j < KK; ++j)
            idxo[((size_t)b * TT + t) * KK + j] = bi[j];
    }
}

// ---- Kernel 6: g[b,o,t,k] = gelu(ynb[b,o,idx[b,t,k]] + yctr[b,o,t]) -------
__global__ void k_gather_gelu(const float* __restrict__ ynb, const float* __restrict__ yctr,
                              const int* __restrict__ idx, float* __restrict__ g) {
    int i = blockIdx.x * blockDim.x + threadIdx.x; // B*C*T
    int t = i % TT;
    int b = i / (TT * CC);
    float yc = yctr[i];
    const int* ip = idx + ((size_t)b * TT + t) * KK;
    const float* yn = ynb + ((size_t)(i / TT)) * TT; // row [b,o,:]
    float* gp = g + (size_t)i * KK;
#pragma unroll
    for (int k = 0; k < KK; ++k)
        gp[k] = gelu_exact(yn[ip[k]] + yc);
}

// ---- Kernel 7: grouped conv1d k=5 pad=2 -> tout ---------------------------
__global__ void k_tconv(const float* __restrict__ h, const float* __restrict__ tw2i,
                        const float* __restrict__ tb2i, float* __restrict__ tout) {
    int i = blockIdx.x * blockDim.x + threadIdx.x; // B*C*T
    int t = i % TT;
    int o = (i / TT) % CC;
    int b = i / (TT * CC);
    int gb = (o / CG) * CG;
    const float* w = tw2i + (size_t)o * CG * 5;
    float s = tb2i[o];
    for (int ci = 0; ci < CG; ++ci) {
        const float* hp = h + ((size_t)b * CC + gb + ci) * TT;
#pragma unroll
        for (int dt = 0; dt < 5; ++dt) {
            int tt2 = t + dt - 2;
            if (tt2 >= 0 && tt2 < TT) s += w[ci * 5 + dt] * hp[tt2];
        }
    }
    tout[i] = s;
}

// ---- Kernel 8: grouped 5x5 conv2d over (t,k), max over k, + residual add --
__global__ void k_dgconv_max_add(const float* __restrict__ g, const float* __restrict__ dw2i,
                                 const float* __restrict__ db2i, const float* __restrict__ tout,
                                 const float* __restrict__ xin, float* __restrict__ xo) {
    int i = blockIdx.x * blockDim.x + threadIdx.x; // B*C*T
    int t = i % TT;
    int o = (i / TT) % CC;
    int b = i / (TT * CC);
    int gb = (o / CG) * CG;
    const float* w = dw2i + (size_t)o * CG * 25;
    float bias = db2i[o];
    float m = -3.0e38f;
    for (int ko = 0; ko < KK; ++ko) {
        float s = bias;
        for (int ci = 0; ci < CG; ++ci) {
            const float* gp = g + ((size_t)b * CC + gb + ci) * TT * KK;
            const float* wc = w + ci * 25;
#pragma unroll
            for (int dt = 0; dt < 5; ++dt) {
                int tt2 = t + dt - 2;
                if (tt2 < 0 || tt2 >= TT) continue;
                const float* gr = gp + (size_t)tt2 * KK;
#pragma unroll
                for (int dk = 0; dk < 5; ++dk) {
                    int kk2 = ko + dk - 2;
                    if (kk2 < 0 || kk2 >= KK) continue;
                    s += wc[dt * 5 + dk] * gr[kk2];
                }
            }
        }
        m = fmaxf(m, s);
    }
    xo[i] = m + tout[i] + xin[i];
}

// ---------------------------------------------------------------------------
extern "C" void kernel_launch(void* const* d_in, const int* in_sizes, int n_in,
                              void* d_out, int out_size, void* d_ws, size_t ws_size,
                              hipStream_t stream) {
    (void)in_sizes; (void)n_in; (void)out_size; (void)ws_size;
    const float* x_in = (const float*)d_in[0];
    const float* tw1  = (const float*)d_in[1];
    const float* tb1  = (const float*)d_in[2];
    const float* tw2  = (const float*)d_in[3];
    const float* tb2  = (const float*)d_in[4];
    const float* dw1  = (const float*)d_in[5];
    const float* db1  = (const float*)d_in[6];
    const float* dw2  = (const float*)d_in[7];
    const float* db2  = (const float*)d_in[8];

    char* p = (char*)d_ws;
    auto take = [&](size_t n) { char* r = p; p += (n + 255) & ~(size_t)255; return (void*)r; };

    unsigned short* xt   = (unsigned short*)take((size_t)BB * TT * CC * 2);
    float*          xx   = (float*)take((size_t)BB * TT * 4);
    unsigned short* wt1  = (unsigned short*)take((size_t)CC * CC * 2);
    unsigned short* wnb  = (unsigned short*)take((size_t)CC * CC * 2);
    unsigned short* wctr = (unsigned short*)take((size_t)CC * CC * 2);
    float* h    = (float*)take((size_t)NEL * 4);
    float* ynb  = (float*)take((size_t)NEL * 4);
    float* yctr = (float*)take((size_t)NEL * 4);
    float* tout = (float*)take((size_t)NEL * 4);
    float* gbuf = (float*)take((size_t)NEL * KK * 4);
    int*   idxb = (int*)take((size_t)BB * TT * KK * 4);
    float* xpings[2] = { (float*)take((size_t)NEL * 4), (float*)take((size_t)NEL * 4) };

    const float* xc = x_in;
    for (int i = 0; i < LL; ++i) {
        float* xo = (i == LL - 1) ? (float*)d_out : xpings[i & 1];
        const float* tw1i = tw1 + (size_t)i * CC * CC;
        const float* tb1i = tb1 + (size_t)i * CC;
        const float* tw2i = tw2 + (size_t)i * CC * CG * 5;
        const float* tb2i = tb2 + (size_t)i * CC;
        const float* dw1i = dw1 + (size_t)i * CC * 2 * CC;
        const float* db1i = db1 + (size_t)i * CC;
        const float* dw2i = dw2 + (size_t)i * CC * CG * 25;
        const float* db2i = db2 + (size_t)i * CC;

        k_transpose_bf16<<<NEL / 256, 256, 0, stream>>>(xc, xt);
        k_norms<<<(BB * TT) / 256, 256, 0, stream>>>(xc, xx);
        k_wcvt<<<(CC * CC) / 256, 256, 0, stream>>>(tw1i, dw1i, wt1, wnb, wctr);

        dim3 ggrid(TT / 128, CC / 64, BB);
        k_gemm<<<ggrid, 256, 0, stream>>>(wt1,  xt, tb1i,    h,    1); // temporal pw + gelu
        k_gemm<<<ggrid, 256, 0, stream>>>(wnb,  xt, nullptr, ynb,  0); // W_nb . x
        k_gemm<<<ggrid, 256, 0, stream>>>(wctr, xt, db1i,    yctr, 0); // W_ctr . x + db1

        k_knn<<<dim3(TT / 16, BB), 32, 0, stream>>>(xt, xx, idxb);
        k_gather_gelu<<<NEL / 256, 256, 0, stream>>>(ynb, yctr, idxb, gbuf);
        k_tconv<<<NEL / 256, 256, 0, stream>>>(h, tw2i, tb2i, tout);
        k_dgconv_max_add<<<NEL / 256, 256, 0, stream>>>(gbuf, dw2i, db2i, tout, xc, xo);
        xc = xo;
    }
}
